// Is_Stable_GNN_72189810312083
// MI455X (gfx1250) — matrix operand order repaired and checked by reading, not compiled, not run on hardware
//
#include <hip/hip_runtime.h>
#include <math.h>

typedef __bf16 bf16_t;
typedef __attribute__((ext_vector_type(16))) __bf16 v16bf;
typedef __attribute__((ext_vector_type(8)))  float  v8f;

// ---------------------------------------------------------------------------
// helpers
// ---------------------------------------------------------------------------
__device__ __forceinline__ float gelu_f(float v) {
  return 0.5f * v * (1.0f + erff(v * 0.70710678118654752f));
}

// A-matrix 16x32 layout (contiguous, matches B swizzle): lane<16 holds K{0..15}, lane>=16 K{16..31}
__device__ __forceinline__ int ka_idx(int lane, int j) {
  return ((lane >> 4) << 4) + j;
}

__device__ __forceinline__ v16bf load_a_frag(const bf16_t* buf, int stride, int m0, int k0, int lane) {
  v16bf a;
  const bf16_t* row = buf + (m0 + (lane & 15)) * stride + k0;
#pragma unroll
  for (int j = 0; j < 16; ++j) a[j] = row[ka_idx(lane, j)];
  return a;
}

// B fragments pre-swizzled in memory: frag*512 + lane*16 + j (fully coalesced b128)
__device__ __forceinline__ v16bf load_b_frag(const bf16_t* frag, int lane) {
  v16bf b;
  const bf16_t* p = frag + lane * 16;
#pragma unroll
  for (int j = 0; j < 16; ++j) b[j] = p[j];
  return b;
}

__device__ __forceinline__ v8f wmma_bf16(v16bf a, v16bf b, v8f c) {
  return __builtin_amdgcn_wmma_f32_16x16x32_bf16(false, a, false, b, (short)0, c, false, false);
}

__device__ __forceinline__ unsigned pack_bf16(float lo, float hi) {
  unsigned short a = __builtin_bit_cast(unsigned short, (bf16_t)lo);
  unsigned short b = __builtin_bit_cast(unsigned short, (bf16_t)hi);
  return (unsigned)a | ((unsigned)b << 16);
}

// ---------------------------------------------------------------------------
// weight pre-swizzle: f32 row-major [K][N] -> bf16 B-fragments
// B layout per 32x16 tile: lane l holds col N=l&15, K = (l>>4)*16 + j
// ---------------------------------------------------------------------------
__global__ void swizzle_weights(const float* __restrict__ w, bf16_t* __restrict__ out,
                                int K, int Nn) {
  int ntiles = Nn >> 4;
  int total  = (K >> 5) * ntiles * 512;
  int t = blockIdx.x * blockDim.x + threadIdx.x;
  if (t >= total) return;
  int frag = t >> 9, r = t & 511, lane = r >> 4, j = r & 15;
  int kt = frag / ntiles, nt = frag % ntiles;
  int k = kt * 32 + ((lane >> 4) << 4) + j;
  int n = nt * 16 + (lane & 15);
  out[t] = (bf16_t)w[k * Nn + n];
}

// ---------------------------------------------------------------------------
// node encoder (tiny; K=92 not a WMMA multiple -> VALU)
// ---------------------------------------------------------------------------
__global__ void encode_nodes(const float* __restrict__ x, const float* __restrict__ w,
                             const float* __restrict__ b, float* __restrict__ out, int N) {
  int t = blockIdx.x * blockDim.x + threadIdx.x;
  if (t >= N * 64) return;
  int row = t >> 6, c = t & 63;
  const float* xr = x + (long)row * 92;
  float s = b[c];
#pragma unroll 4
  for (int k = 0; k < 92; ++k) s += xr[k] * w[k * 64 + c];
  out[t] = s;
}

// ---------------------------------------------------------------------------
// fused edge encoder: pass 1 computes global-LN stats on the fly (recompute is
// ~free vs. 800 MB of f32 round-trips), pass 2 re-encodes + applies LN and
// writes bf16 directly (non-temporal: streamed, keep L2 for the node table)
// ---------------------------------------------------------------------------
__global__ void encode_edges_stats(const float* __restrict__ ea, const float* __restrict__ w,
                                   const float* __restrict__ b, int E,
                                   float* __restrict__ partials) {
  __shared__ float s1[256], s2[256];
  int tid = threadIdx.x;
  float a = 0.f, q = 0.f;
  int total = E * 128;
  for (int t = blockIdx.x * 256 + tid; t < total; t += gridDim.x * 256) {
    int row = t >> 7, c = t & 127;
    const float* er = ea + (long)row * 50;
    float s = b[c];
#pragma unroll 5
    for (int k = 0; k < 50; ++k) s += er[k] * w[k * 128 + c];
    a += s; q += s * s;
  }
  s1[tid] = a; s2[tid] = q; __syncthreads();
  for (int off = 128; off > 0; off >>= 1) {
    if (tid < off) { s1[tid] += s1[tid + off]; s2[tid] += s2[tid + off]; }
    __syncthreads();
  }
  if (tid == 0) { partials[blockIdx.x] = s1[0]; partials[gridDim.x + blockIdx.x] = s2[0]; }
}

__global__ void encode_edges_ln_bf16(const float* __restrict__ ea, const float* __restrict__ w,
                                     const float* __restrict__ b, const float* __restrict__ lnw,
                                     const float* __restrict__ lnb, const float* __restrict__ stats,
                                     unsigned* __restrict__ out, int E) {
  int t = blockIdx.x * blockDim.x + threadIdx.x;  // one thread = 2 channels
  if (t >= E * 64) return;
  int row = t >> 6, cp = t & 63;
  int c0 = cp * 2, c1 = c0 + 1;
  const float* er = ea + (long)row * 50;
  float s0 = b[c0], s1v = b[c1];
#pragma unroll 5
  for (int k = 0; k < 50; ++k) {
    float e = er[k];
    s0  += e * w[k * 128 + c0];
    s1v += e * w[k * 128 + c1];
  }
  float mean = stats[0], inv = stats[1];
  float v0 = (s0  - mean) * inv * lnw[c0] + lnb[c0];
  float v1 = (s1v - mean) * inv * lnw[c1] + lnb[c1];
  __builtin_nontemporal_store(pack_bf16(v0, v1), &out[(long)row * 64 + cp]);
}

// ---------------------------------------------------------------------------
// graph LayerNorm over f32 buffers (deterministic 2-stage)
// ---------------------------------------------------------------------------
__global__ void reduce_stats(const float* __restrict__ x, int n, float* __restrict__ partials) {
  __shared__ float s1[256], s2[256];
  int tid = threadIdx.x;
  float a = 0.f, b = 0.f;
  for (int i = blockIdx.x * 256 + tid; i < n; i += gridDim.x * 256) {
    float v = x[i]; a += v; b += v * v;
  }
  s1[tid] = a; s2[tid] = b; __syncthreads();
  for (int off = 128; off > 0; off >>= 1) {
    if (tid < off) { s1[tid] += s1[tid + off]; s2[tid] += s2[tid + off]; }
    __syncthreads();
  }
  if (tid == 0) { partials[blockIdx.x] = s1[0]; partials[gridDim.x + blockIdx.x] = s2[0]; }
}

__global__ void finalize_stats(const float* __restrict__ partials, int G, float count,
                               float* __restrict__ stats) {
  __shared__ float s1[256], s2[256];
  int tid = threadIdx.x;
  float a = 0.f, b = 0.f;
  for (int i = tid; i < G; i += 256) { a += partials[i]; b += partials[G + i]; }
  s1[tid] = a; s2[tid] = b; __syncthreads();
  for (int off = 128; off > 0; off >>= 1) {
    if (tid < off) { s1[tid] += s1[tid + off]; s2[tid] += s2[tid + off]; }
    __syncthreads();
  }
  if (tid == 0) {
    float mean = s1[0] / count;
    float var  = s2[0] / count - mean * mean;
    stats[0] = mean;
    stats[1] = rsqrtf(var + 1e-5f);
  }
}

__global__ void apply_ln(const float* __restrict__ in, float* __restrict__ out,
                         const float* __restrict__ res, const float* __restrict__ stats,
                         const float* __restrict__ w, const float* __restrict__ b,
                         int n, int cmask) {
  int t = blockIdx.x * blockDim.x + threadIdx.x;
  if (t >= n) return;
  int c = t & cmask;
  float v = (in[t] - stats[0]) * stats[1] * w[c] + b[c];
  if (res) v += res[t];
  out[t] = v;
}

// ---------------------------------------------------------------------------
// main per-layer edge kernel: 1 wave = 32 edges, all GEMMs on WMMA bf16
// ---------------------------------------------------------------------------
#define TE 32

__global__ __launch_bounds__(32)
void edge_layer(const float* __restrict__ x, const unsigned* __restrict__ ea2,
                const int* __restrict__ src, const int* __restrict__ dst,
                const bf16_t* __restrict__ WfS, const float* __restrict__ bf_,
                const bf16_t* __restrict__ WsS, const float* __restrict__ bs_,
                const bf16_t* __restrict__ WqS, const float* __restrict__ bq_,
                const bf16_t* __restrict__ WoS, const float* __restrict__ bo_,
                float* __restrict__ agg, int E) {
  __shared__ bf16_t zt[TE * 256];    // z = [x_dst | x_src | ea]
  __shared__ bf16_t msgb[TE * 64];   // msg, later reused for attention output
  __shared__ bf16_t qkvb[TE * 192];  // qkv
  __shared__ float  ot[TE * 64];     // final per-edge output
  __shared__ int    sidx[2 * TE];

  const int lane = threadIdx.x;
  const int e0   = blockIdx.x * TE;

  {
    int e = e0 + lane;
    sidx[lane]      = (e < E) ? dst[e] : -1;
    sidx[TE + lane] = (e < E) ? src[e] : -1;
  }
  __syncthreads();

  // gather z into LDS: x rows (f32->bf16, L2-resident), ea rows (bf16 copy, NT)
  for (int t = lane; t < TE * 64; t += 32) {
    int e = t >> 6, c = t & 63;
    int di = sidx[e], si = sidx[TE + e];
    zt[e * 256 + c]      = (di >= 0) ? (bf16_t)x[di * 64 + c] : (bf16_t)0.0f;
    zt[e * 256 + 64 + c] = (si >= 0) ? (bf16_t)x[si * 64 + c] : (bf16_t)0.0f;
  }
  for (int t = lane; t < TE * 64; t += 32) {  // 2 bf16 per 32-bit move
    int e = t >> 6, c = t & 63;
    unsigned v = (e0 + e < E) ? __builtin_nontemporal_load(&ea2[(long)(e0 + e) * 64 + c]) : 0u;
    *(unsigned*)&zt[e * 256 + 128 + 2 * c] = v;
  }
  __syncthreads();

  const int mrow = lane & 15;  // col within D tile / row within A tile
  const int hi   = lane >> 4;

  // ---- gated message: msg = gelu(z@Wf+bf) * gelu(z@Ws+bs)  [32x256 @ 256x64]
  for (int nt = 0; nt < 4; ++nt) {
    v8f cf0 = {}, cf1 = {}, cs0 = {}, cs1 = {};
#pragma unroll
    for (int kt = 0; kt < 8; ++kt) {
      v16bf a0  = load_a_frag(zt, 256, 0,  kt * 32, lane);
      v16bf a1  = load_a_frag(zt, 256, 16, kt * 32, lane);
      v16bf bfm = load_b_frag(WfS + (kt * 4 + nt) * 512, lane);
      v16bf bsm = load_b_frag(WsS + (kt * 4 + nt) * 512, lane);
      cf0 = wmma_bf16(a0, bfm, cf0); cf1 = wmma_bf16(a1, bfm, cf1);
      cs0 = wmma_bf16(a0, bsm, cs0); cs1 = wmma_bf16(a1, bsm, cs1);
    }
    int n = nt * 16 + mrow;
    float bfv = bf_[n], bsv = bs_[n];
#pragma unroll
    for (int r = 0; r < 8; ++r) {
      int m0 = hi * 8 + r;
      msgb[m0 * 64 + n]        = (bf16_t)(gelu_f(cf0[r] + bfv) * gelu_f(cs0[r] + bsv));
      msgb[(16 + m0) * 64 + n] = (bf16_t)(gelu_f(cf1[r] + bfv) * gelu_f(cs1[r] + bsv));
    }
  }
  __syncthreads();

  // ---- qkv = msg @ Wqkv + b   [32x64 @ 64x192]
  for (int nt = 0; nt < 12; ++nt) {
    v8f c0 = {}, c1 = {};
#pragma unroll
    for (int kt = 0; kt < 2; ++kt) {
      v16bf a0 = load_a_frag(msgb, 64, 0,  kt * 32, lane);
      v16bf a1 = load_a_frag(msgb, 64, 16, kt * 32, lane);
      v16bf b  = load_b_frag(WqS + (kt * 12 + nt) * 512, lane);
      c0 = wmma_bf16(a0, b, c0); c1 = wmma_bf16(a1, b, c1);
    }
    int n = nt * 16 + mrow;
    float bv = bq_[n];
#pragma unroll
    for (int r = 0; r < 8; ++r) {
      int m0 = hi * 8 + r;
      qkvb[m0 * 192 + n]        = (bf16_t)(c0[r] + bv);
      qkvb[(16 + m0) * 192 + n] = (bf16_t)(c1[r] + bv);
    }
  }
  __syncthreads();

  // ---- 4x4 head-attention, one edge per lane; per head h: q@h*48, k@h*48+16, v@h*48+32
  {
    const bf16_t* q = qkvb + lane * 192;
    const float scale = 0.25f;  // HEAD_DIM^-0.5
    for (int h = 0; h < 4; ++h) {
      float att[4]; float mx = -1e30f;
#pragma unroll
      for (int g = 0; g < 4; ++g) {
        float s = 0.f;
#pragma unroll
        for (int d = 0; d < 16; ++d) s += (float)q[h * 48 + d] * (float)q[g * 48 + 16 + d];
        s *= scale; att[g] = s; mx = fmaxf(mx, s);
      }
      float den = 0.f;
#pragma unroll
      for (int g = 0; g < 4; ++g) { att[g] = __expf(att[g] - mx); den += att[g]; }
      float inv = 1.f / den;
#pragma unroll
      for (int d = 0; d < 16; ++d) {
        float acc = 0.f;
#pragma unroll
        for (int g = 0; g < 4; ++g) acc += att[g] * (float)q[g * 48 + 32 + d];
        msgb[lane * 64 + h * 16 + d] = (bf16_t)(acc * inv);  // reuse msgb as o-buffer
      }
    }
  }
  __syncthreads();

  // ---- o = o @ Wout + b  [32x64 @ 64x64]
  for (int nt = 0; nt < 4; ++nt) {
    v8f c0 = {}, c1 = {};
#pragma unroll
    for (int kt = 0; kt < 2; ++kt) {
      v16bf a0 = load_a_frag(msgb, 64, 0,  kt * 32, lane);
      v16bf a1 = load_a_frag(msgb, 64, 16, kt * 32, lane);
      v16bf b  = load_b_frag(WoS + (kt * 4 + nt) * 512, lane);
      c0 = wmma_bf16(a0, b, c0); c1 = wmma_bf16(a1, b, c1);
    }
    int n = nt * 16 + mrow;
    float bv = bo_[n];
#pragma unroll
    for (int r = 0; r < 8; ++r) {
      int m0 = hi * 8 + r;
      ot[m0 * 64 + n]        = c0[r] + bv;
      ot[(16 + m0) * 64 + n] = c1[r] + bv;
    }
  }
  __syncthreads();

  // ---- scatter-add into agg (L2-resident)
  for (int t = lane; t < TE * 64; t += 32) {
    int e = t >> 6, c = t & 63;
    int di = sidx[e];
    if (di >= 0) atomicAdd(&agg[(long)di * 64 + c], ot[t]);
  }
}

// ---------------------------------------------------------------------------
// tail: 64->64 gelu MLP (rows = N or 64), pooling, head
// ---------------------------------------------------------------------------
__global__ void mlp64(const float* __restrict__ in, const float* __restrict__ w,
                      const float* __restrict__ b, float* __restrict__ out, int rows) {
  int t = blockIdx.x * blockDim.x + threadIdx.x;
  if (t >= rows * 64) return;
  int row = t >> 6, c = t & 63;
  const float* xr = in + (long)row * 64;
  float s = b[c];
#pragma unroll 8
  for (int k = 0; k < 64; ++k) s += xr[k] * w[k * 64 + c];
  out[t] = gelu_f(s);
}

__global__ void pool_sum(const float* __restrict__ x, const int* __restrict__ batch,
                         float* __restrict__ pooled, float* __restrict__ cnt, int N) {
  int t = blockIdx.x * blockDim.x + threadIdx.x;
  if (t >= N * 64) return;
  int row = t >> 6, c = t & 63;
  int g = batch[row];
  atomicAdd(&pooled[g * 64 + c], x[t]);
  if (c == 0) atomicAdd(&cnt[g], 1.0f);
}

__global__ void pool_div(float* __restrict__ pooled, const float* __restrict__ cnt) {
  int t = blockIdx.x * blockDim.x + threadIdx.x;
  if (t >= 64 * 64) return;
  pooled[t] /= fmaxf(cnt[t >> 6], 1.0f);
}

__global__ void head_k(const float* __restrict__ g, const float* __restrict__ w,
                       const float* __restrict__ b, float* __restrict__ out) {
  int r = threadIdx.x;
  if (r >= 64) return;
  float s = b[0];
#pragma unroll 8
  for (int k = 0; k < 64; ++k) s += g[r * 64 + k] * w[k];
  out[r] = s;
}

// ---------------------------------------------------------------------------
// orchestration
// ---------------------------------------------------------------------------
extern "C" void kernel_launch(void* const* d_in, const int* in_sizes, int n_in,
                              void* d_out, int out_size, void* d_ws, size_t ws_size,
                              hipStream_t stream) {
  const float* x_in    = (const float*)d_in[0];
  const int*   eidx    = (const int*)d_in[1];
  const float* eattr   = (const float*)d_in[2];
  const int*   batch   = (const int*)d_in[3];
  const float* enc_a_w = (const float*)d_in[4];
  const float* enc_a_b = (const float*)d_in[5];
  const float* enc_e_w = (const float*)d_in[6];
  const float* enc_e_b = (const float*)d_in[7];
  const float* ln1_w   = (const float*)d_in[8];
  const float* ln1_b   = (const float*)d_in[9];
  const float* ln2_w   = (const float*)d_in[10];
  const float* ln2_b   = (const float*)d_in[11];
  const float* lin_f_w = (const float*)d_in[12];
  const float* lin_f_b = (const float*)d_in[13];
  const float* lin_s_w = (const float*)d_in[14];
  const float* lin_s_b = (const float*)d_in[15];
  const float* qkv_w   = (const float*)d_in[16];
  const float* qkv_b   = (const float*)d_in[17];
  const float* outp_w  = (const float*)d_in[18];
  const float* outp_b  = (const float*)d_in[19];
  const float* bn_w    = (const float*)d_in[20];
  const float* bn_b    = (const float*)d_in[21];
  const float* l1_w    = (const float*)d_in[22];
  const float* l1_b    = (const float*)d_in[23];
  const float* l2_w    = (const float*)d_in[24];
  const float* l2_b    = (const float*)d_in[25];
  const float* head_w  = (const float*)d_in[26];
  const float* head_b  = (const float*)d_in[27];

  const int N = in_sizes[0] / 92;
  const int E = in_sizes[1] / 2;
  const int* srcIdx = eidx;      // edge_index[0]
  const int* dstIdx = eidx + E;  // edge_index[1]

  char* wsp = (char*)d_ws;
  auto alloc = [&](size_t bytes) -> void* {
    void* p = (void*)wsp;
    wsp += (bytes + 255) & ~(size_t)255;
    return p;
  };
  float*    xA       = (float*)alloc((size_t)N * 64 * 4);
  float*    xB       = (float*)alloc((size_t)N * 64 * 4);
  float*    agg      = (float*)alloc((size_t)N * 64 * 4);
  unsigned* eaH      = (unsigned*)alloc((size_t)E * 64 * 4);  // bf16 pairs [E,128]
  bf16_t*   WfS      = (bf16_t*)alloc(16384 * 2);
  bf16_t*   WsS      = (bf16_t*)alloc(16384 * 2);
  bf16_t*   WqS      = (bf16_t*)alloc(12288 * 2);
  bf16_t*   WoS      = (bf16_t*)alloc(4096 * 2);
  float*    partials = (float*)alloc(2 * 2048 * 4);
  float*    stats    = (float*)alloc(256);
  float*    pooled   = (float*)alloc(64 * 64 * 4);
  float*    cnt      = (float*)alloc(64 * 4);
  float*    gbuf     = (float*)alloc(64 * 64 * 4);

  // pre-swizzle weights into WMMA B-fragment order (bf16, stays hot in L2)
  swizzle_weights<<<(16384 + 255) / 256, 256, 0, stream>>>(lin_f_w, WfS, 256, 64);
  swizzle_weights<<<(16384 + 255) / 256, 256, 0, stream>>>(lin_s_w, WsS, 256, 64);
  swizzle_weights<<<(12288 + 255) / 256, 256, 0, stream>>>(qkv_w,   WqS, 64, 192);
  swizzle_weights<<<(4096 + 255) / 256, 256, 0, stream>>>(outp_w,  WoS, 64, 64);

  auto run_ln = [&](const float* in, float* out, const float* res,
                    const float* w, const float* b, int n, int C) {
    reduce_stats<<<1024, 256, 0, stream>>>(in, n, partials);
    finalize_stats<<<1, 256, 0, stream>>>(partials, 1024, (float)n, stats);
    apply_ln<<<(n + 255) / 256, 256, 0, stream>>>(in, out, res, stats, w, b, n, C - 1);
  };

  // node encoder + graph LN
  encode_nodes<<<(N * 64 + 255) / 256, 256, 0, stream>>>(x_in, enc_a_w, enc_a_b, xA, N);
  run_ln(xA, xA, nullptr, ln1_w, ln1_b, N * 64, 64);

  // fused edge encoder + graph LN -> bf16 (single HBM write of ea, no f32 buffer)
  encode_edges_stats<<<2048, 256, 0, stream>>>(eattr, enc_e_w, enc_e_b, E, partials);
  finalize_stats<<<1, 256, 0, stream>>>(partials, 2048, (float)E * 128.f, stats);
  encode_edges_ln_bf16<<<(E * 64 + 255) / 256, 256, 0, stream>>>(
      eattr, enc_e_w, enc_e_b, ln2_w, ln2_b, stats, eaH, E);

  // 4 message-passing layers
  for (int layer = 0; layer < 4; ++layer) {
    hipMemsetAsync(agg, 0, (size_t)N * 64 * 4, stream);
    edge_layer<<<(E + TE - 1) / TE, 32, 0, stream>>>(
        xA, eaH, srcIdx, dstIdx,
        WfS, lin_f_b, WsS, lin_s_b, WqS, qkv_b, WoS, outp_b, agg, E);
    run_ln(agg, xA, xA, bn_w, bn_b, N * 64, 64);  // x = LN(agg) + x
  }

  // node MLP (4 layers, ping-pong, ends in xA)
  {
    float* pin = xA; float* pout = xB;
    for (int i = 0; i < 4; ++i) {
      mlp64<<<(N * 64 + 255) / 256, 256, 0, stream>>>(pin, l1_w + i * 4096, l1_b + i * 64, pout, N);
      float* t = pin; pin = pout; pout = t;
    }
  }

  // mean pooling per graph
  hipMemsetAsync(pooled, 0, 64 * 64 * 4, stream);
  hipMemsetAsync(cnt, 0, 64 * 4, stream);
  pool_sum<<<(N * 64 + 255) / 256, 256, 0, stream>>>(xA, batch, pooled, cnt, N);
  pool_div<<<16, 256, 0, stream>>>(pooled, cnt);

  // graph MLP (4 layers, ping-pong pooled<->gbuf, ends in pooled)
  {
    float* pin = pooled; float* pout = gbuf;
    for (int i = 0; i < 4; ++i) {
      mlp64<<<16, 256, 0, stream>>>(pin, l2_w + i * 4096, l2_b + i * 64, pout, 64);
      float* t = pin; pin = pout; pout = t;
    }
    head_k<<<1, 64, 0, stream>>>(pin, head_w, head_b, (float*)d_out);
  }
}
